// decoder_66546223284402
// MI455X (gfx1250) — matrix operand order repaired and checked by reading, not compiled, not run on hardware
//
#include <hip/hip_runtime.h>

// ---------------- problem constants (from reference) ----------------
#define BB      128
#define TVv     128
#define TAa     128
#define LL      16
#define CTXC    512
#define HH      1024
#define EVv     1024
#define EAa     1024
#define VOCABV  20000
#define INFEAT  1024

// ---------------- vector types ----------------
typedef __attribute__((ext_vector_type(16))) __bf16 v16bf;
typedef __attribute__((ext_vector_type(8)))  __bf16 v8bf;
typedef __attribute__((ext_vector_type(8)))  float  v8f;
typedef __attribute__((ext_vector_type(8)))  float  f32x8;

// ---------------- helpers ----------------
__device__ __forceinline__ v8f wmma_bf16(v16bf a, v16bf b, v8f c) {
    return __builtin_amdgcn_wmma_f32_16x16x32_bf16(
        /*neg_a=*/false, a, /*neg_b=*/false, b,
        /*c_mod=*/(short)0, c, /*reuse_a=*/false, /*reuse_b=*/false);
}

// Per-lane 32-bit element offsets (uniform base + voffset addressing).
// A fragment (16x32 bf16): lane l holds row m0+(l&15), k = (i>>3)*16 + (l>>4)*8 + (i&7)
__device__ __forceinline__ unsigned a_off(long lda, int m0, int lane, int M) {
    int m = m0 + (lane & 15); if (m >= M) m = M - 1;
    return (unsigned)((long)m * lda) + (unsigned)((lane >> 4) << 3);
}
// B fragment (32x16 bf16) from row-major W[N,K]: lane l holds col n0+(l&15),
// k = (i>>3)*16 + (l>>4)*8 + (i&7)
__device__ __forceinline__ unsigned b_off(long ldw, int n0, int lane, int N) {
    int n = n0 + (lane & 15); if (n >= N) n = N - 1;
    return (unsigned)((long)n * ldw) + (unsigned)((lane >> 4) << 3);
}
__device__ __forceinline__ v16bf load_a_at(const __bf16* __restrict__ A, unsigned off) {
    v8bf lo = *(const v8bf*)(A + off);
    v8bf hi = *(const v8bf*)(A + off + 16);
    return __builtin_shufflevector(lo, hi, 0,1,2,3,4,5,6,7,8,9,10,11,12,13,14,15);
}
__device__ __forceinline__ v16bf load_b_at(const __bf16* __restrict__ W, unsigned off) {
    v8bf lo = *(const v8bf*)(W + off);
    v8bf hi = *(const v8bf*)(W + off + 16);
    return __builtin_shufflevector(lo, hi, 0,1,2,3,4,5,6,7,8,9,10,11,12,13,14,15);
}

__device__ __forceinline__ float sigmf(float x) { return 1.0f / (1.0f + expf(-x)); }

// ---------------- conversion kernels (streaming, vectorized 8-wide) ----------------
__global__ void cvt_f32_bf16(const float* __restrict__ s, __bf16* __restrict__ d, long n) {
    long i = ((long)blockIdx.x * blockDim.x + threadIdx.x) * 8;
    long stride = (long)gridDim.x * blockDim.x * 8;
    for (; i < n; i += stride) {
        f32x8 v = *(const f32x8*)(s + i);
        v8bf o;
#pragma unroll
        for (int j = 0; j < 8; ++j) o[j] = (__bf16)v[j];
        *(v8bf*)(d + i) = o;
    }
}

// strided 2D convert: rows = gridDim.x
__global__ void cvt2d_f32_bf16(const float* __restrict__ s, long srow,
                               __bf16* __restrict__ d, long drow, int cols) {
    long r = blockIdx.x;
    for (int c = threadIdx.x * 8; c < cols; c += blockDim.x * 8) {
        f32x8 v = *(const f32x8*)(s + r * srow + c);
        v8bf o;
#pragma unroll
        for (int j = 0; j < 8; ++j) o[j] = (__bf16)v[j];
        *(v8bf*)(d + r * drow + c) = o;
    }
}

// hidden_states [L,B,H] fp32 -> [B,L,H] bf16
__global__ void cvt_hs(const float* __restrict__ hs, __bf16* __restrict__ d) {
    int lb = blockIdx.x; int l = lb / BB; int b = lb % BB;
    const float* src = hs + ((long)l * BB + b) * HH;
    __bf16* dst = d + ((long)b * LL + l) * HH;
    for (int h = threadIdx.x * 8; h < HH; h += blockDim.x * 8) {
        f32x8 v = *(const f32x8*)(src + h);
        v8bf o;
#pragma unroll
        for (int j = 0; j < 8; ++j) o[j] = (__bf16)v[j];
        *(v8bf*)(dst + h) = o;
    }
}

// ---------------- generic WMMA GEMM: D = A @ W^T (+bias1+bias2) ----------------
// bf16 A [M,K] lda, bf16 W [N,K] ldw, fp32 D [M,ldd]; 256 thr = 8 waves;
// wave tile 32x64 (2x4 wmma accumulators), block tile 128x128.
// Ping-pong pipelined K-loop (two k-steps per iteration, no buffer rotation).
// Requires K % 64 == 0 (all call sites: 512/1024/1536).
__global__ void __launch_bounds__(256, 2)
gemm_bf16_wmma(const __bf16* __restrict__ A, long lda,
               const __bf16* __restrict__ W, long ldw,
               const float* __restrict__ bias1,
               const float* __restrict__ bias2,
               float* __restrict__ D, long ldd,
               int M, int N, int K, int accumulate) {
    const int lane = threadIdx.x & 31;
    const int wv   = threadIdx.x >> 5;
    const int m0   = blockIdx.x * 128 + (wv & 3) * 32;
    const int n0   = blockIdx.y * 128 + (wv >> 2) * 64;
    if (m0 >= M) return;                       // wave-uniform
    v8f acc[2][4] = {};

    const unsigned oa0 = a_off(lda, m0,      lane, M);
    const unsigned oa1 = a_off(lda, m0 + 16, lane, M);
    unsigned ob[4];
#pragma unroll
    for (int tj = 0; tj < 4; ++tj) ob[tj] = b_off(ldw, n0 + tj * 16, lane, N);

    // buffer set A holds the "current" k-step fragments
    v16bf a0A = load_a_at(A, oa0);
    v16bf a1A = load_a_at(A, oa1);
    v16bf bA[4];
#pragma unroll
    for (int tj = 0; tj < 4; ++tj) bA[tj] = load_b_at(W, ob[tj]);
    v16bf a0B, a1B, bB[4];

    int k0 = 32;
    for (; k0 < K - 32; k0 += 64) {
        // load set B for k0 while computing set A (k0-32)
        a0B = load_a_at(A, oa0 + k0);
        a1B = load_a_at(A, oa1 + k0);
#pragma unroll
        for (int tj = 0; tj < 4; ++tj) bB[tj] = load_b_at(W, ob[tj] + k0);
#pragma unroll
        for (int tj = 0; tj < 4; ++tj) {
            acc[0][tj] = wmma_bf16(a0A, bA[tj], acc[0][tj]);
            acc[1][tj] = wmma_bf16(a1A, bA[tj], acc[1][tj]);
        }
        // load set A for k0+32 while computing set B (k0)
        a0A = load_a_at(A, oa0 + k0 + 32);
        a1A = load_a_at(A, oa1 + k0 + 32);
#pragma unroll
        for (int tj = 0; tj < 4; ++tj) bA[tj] = load_b_at(W, ob[tj] + k0 + 32);
#pragma unroll
        for (int tj = 0; tj < 4; ++tj) {
            acc[0][tj] = wmma_bf16(a0B, bB[tj], acc[0][tj]);
            acc[1][tj] = wmma_bf16(a1B, bB[tj], acc[1][tj]);
        }
    }
    if (k0 < K) {                              // tail pair: k0 == K-32
        a0B = load_a_at(A, oa0 + k0);
        a1B = load_a_at(A, oa1 + k0);
#pragma unroll
        for (int tj = 0; tj < 4; ++tj) bB[tj] = load_b_at(W, ob[tj] + k0);
#pragma unroll
        for (int tj = 0; tj < 4; ++tj) {
            acc[0][tj] = wmma_bf16(a0A, bA[tj], acc[0][tj]);
            acc[1][tj] = wmma_bf16(a1A, bA[tj], acc[1][tj]);
        }
#pragma unroll
        for (int tj = 0; tj < 4; ++tj) {
            acc[0][tj] = wmma_bf16(a0B, bB[tj], acc[0][tj]);
            acc[1][tj] = wmma_bf16(a1B, bB[tj], acc[1][tj]);
        }
    } else {                                   // K == 32
#pragma unroll
        for (int tj = 0; tj < 4; ++tj) {
            acc[0][tj] = wmma_bf16(a0A, bA[tj], acc[0][tj]);
            acc[1][tj] = wmma_bf16(a1A, bA[tj], acc[1][tj]);
        }
    }

    const int half = lane >> 4, nl = lane & 15;
#pragma unroll
    for (int ti = 0; ti < 2; ++ti)
#pragma unroll
        for (int tj = 0; tj < 4; ++tj) {
            int n = n0 + tj * 16 + nl;
            if (n >= N) continue;
            float bs = (bias1 ? bias1[n] : 0.0f) + (bias2 ? bias2[n] : 0.0f);
#pragma unroll
            for (int r = 0; r < 8; ++r) {
                int m = m0 + ti * 16 + half * 8 + r;
                if (m >= M) continue;
                float v = acc[ti][tj][r] + bs;
                if (accumulate) v += D[(long)m * ldd + n];
                D[(long)m * ldd + n] = v;
            }
        }
}

// ---------------- fused attention scores (the hot kernel) ----------------
// X [M,K] bf16 (keys, M = B*T), Wt [E,K] bf16, v1 [B,E] fp32, w2 [E] fp32.
// scores[m] = sum_e tanh( (X@Wt^T)[m,e] + v1[m/T, e] ) * w2[e]
// 8 waves/block, 16 rows/wave; 4 n-tiles per A-fragment; ping-pong pipelined
// K-loop (K % 64 == 0); v2 never materialized.
__global__ void __launch_bounds__(256, 2)
attn_scores_wmma(const __bf16* __restrict__ X,
                 const __bf16* __restrict__ Wt,
                 const float* __restrict__ v1,
                 const float* __restrict__ w2,
                 float* __restrict__ scores,
                 int M, int T, int K, int E) {
    const int lane = threadIdx.x & 31;
    const int wv   = threadIdx.x >> 5;
    const int m0   = blockIdx.x * 128 + wv * 16;
    if (m0 >= M) return;                       // wave-uniform
    const int half = lane >> 4, nl = lane & 15;
    long bofs[8];
#pragma unroll
    for (int r = 0; r < 8; ++r) bofs[r] = (long)((m0 + half * 8 + r) / T) * E;
    const unsigned oa = a_off(K, m0, lane, M);
    float part[8] = {};
    for (int n0 = 0; n0 < E; n0 += 64) {
        v8f acc[4] = {};
        unsigned ob[4];
#pragma unroll
        for (int j = 0; j < 4; ++j) ob[j] = b_off(K, n0 + j * 16, lane, E);

        v16bf aA = load_a_at(X, oa);
        v16bf bA[4];
#pragma unroll
        for (int j = 0; j < 4; ++j) bA[j] = load_b_at(Wt, ob[j]);
        v16bf aB, bB[4];

        int k0 = 32;
        for (; k0 < K - 32; k0 += 64) {
            aB = load_a_at(X, oa + k0);
#pragma unroll
            for (int j = 0; j < 4; ++j) bB[j] = load_b_at(Wt, ob[j] + k0);
#pragma unroll
            for (int j = 0; j < 4; ++j) acc[j] = wmma_bf16(aA, bA[j], acc[j]);
            aA = load_a_at(X, oa + k0 + 32);
#pragma unroll
            for (int j = 0; j < 4; ++j) bA[j] = load_b_at(Wt, ob[j] + k0 + 32);
#pragma unroll
            for (int j = 0; j < 4; ++j) acc[j] = wmma_bf16(aB, bB[j], acc[j]);
        }
        if (k0 < K) {                          // tail pair: k0 == K-32
            aB = load_a_at(X, oa + k0);
#pragma unroll
            for (int j = 0; j < 4; ++j) bB[j] = load_b_at(Wt, ob[j] + k0);
#pragma unroll
            for (int j = 0; j < 4; ++j) acc[j] = wmma_bf16(aA, bA[j], acc[j]);
#pragma unroll
            for (int j = 0; j < 4; ++j) acc[j] = wmma_bf16(aB, bB[j], acc[j]);
        } else {                               // K == 32
#pragma unroll
            for (int j = 0; j < 4; ++j) acc[j] = wmma_bf16(aA, bA[j], acc[j]);
        }

#pragma unroll
        for (int j = 0; j < 4; ++j) {
            int n = n0 + j * 16 + nl;
            float wvn = w2[n];
#pragma unroll
            for (int r = 0; r < 8; ++r)
                part[r] += tanhf(acc[j][r] + v1[bofs[r] + n]) * wvn;
        }
    }
#pragma unroll
    for (int r = 0; r < 8; ++r) {
        float s = part[r];
        s += __shfl_xor(s, 1);
        s += __shfl_xor(s, 2);
        s += __shfl_xor(s, 4);
        s += __shfl_xor(s, 8);                 // reduce the 16 lanes sharing a row
        if (nl == 0) scores[m0 + half * 8 + r] = s;
    }
}

// ---------------- softmax over T + weighted sum of values ----------------
__global__ void softmax_ctx(const float* __restrict__ scores,
                            const float* __restrict__ X, long strideB, long strideT,
                            float* __restrict__ ctx, int T, int E) {
    int b = blockIdx.x;
    __shared__ float sm[128];
    for (int t = threadIdx.x; t < T; t += blockDim.x) sm[t] = scores[(long)b * T + t];
    __syncthreads();
    if (threadIdx.x == 0) {
        float mx = sm[0];
        for (int t = 1; t < T; ++t) mx = fmaxf(mx, sm[t]);
        float sum = 0.f;
        for (int t = 0; t < T; ++t) { float e = expf(sm[t] - mx); sm[t] = e; sum += e; }
        float inv = 1.0f / sum;
        for (int t = 0; t < T; ++t) sm[t] *= inv;
    }
    __syncthreads();
    const float* Xb = X + (long)b * strideB;
    for (int e = threadIdx.x; e < E; e += blockDim.x) {
        float a = 0.f;
        for (int t = 0; t < T; ++t) a += sm[t] * Xb[(long)t * strideT + e];
        ctx[(long)b * E + e] = a;
    }
}

// ---------------- LSTM pointwise ----------------
__global__ void lstm_cell(const float* __restrict__ gates, const float* __restrict__ c_prev,
                          float* __restrict__ h_out, float* __restrict__ c_out,
                          __bf16* __restrict__ h_bf) {
    int b = blockIdx.x;
    const float* g = gates + (long)b * 4 * HH;
    for (int h = threadIdx.x; h < HH; h += blockDim.x) {
        float i  = sigmf(g[h]);
        float f  = sigmf(g[HH + h]);
        float gg = tanhf(g[2 * HH + h]);
        float o  = sigmf(g[3 * HH + h]);
        float c  = f * c_prev[(long)b * HH + h] + i * gg;
        float hn = o * tanhf(c);
        c_out[(long)b * HH + h] = c;
        h_out[(long)b * HH + h] = hn;
        h_bf[(long)b * HH + h]  = (__bf16)hn;
    }
}

// ---------------- host launcher ----------------
extern "C" void kernel_launch(void* const* d_in, const int* in_sizes, int n_in,
                              void* d_out, int out_size, void* d_ws, size_t ws_size,
                              hipStream_t stream) {
    (void)in_sizes; (void)n_in; (void)out_size; (void)ws_size;

    const float* input   = (const float*)d_in[0];
    const float* visual  = (const float*)d_in[1];
    const float* audio   = (const float*)d_in[2];
    const float* hstates = (const float*)d_in[3];
    const float* cell    = (const float*)d_in[4];
    const float* ctxq    = (const float*)d_in[5];
    const float* Wv0 = (const float*)d_in[6],  *Wv1 = (const float*)d_in[7],  *Wv2 = (const float*)d_in[8];
    const float* Wa0 = (const float*)d_in[9],  *Wa1 = (const float*)d_in[10], *Wa2 = (const float*)d_in[11];
    const float* Wh0 = (const float*)d_in[12], *Wh1 = (const float*)d_in[13], *Wh2 = (const float*)d_in[14];
    const float* Wc0 = (const float*)d_in[15], *Wc1 = (const float*)d_in[16], *Wc2 = (const float*)d_in[17];
    const float* Wac = (const float*)d_in[18], *bac = (const float*)d_in[19];
    const float* Wvc = (const float*)d_in[20], *bvc = (const float*)d_in[21];
    const float* Whc = (const float*)d_in[22], *bhc = (const float*)d_in[23];
    const float* W_ih = (const float*)d_in[24], *W_hh = (const float*)d_in[25];
    const float* b_ih = (const float*)d_in[26], *b_hh = (const float*)d_in[27];
    const float* W_out = (const float*)d_in[28], *b_out = (const float*)d_in[29];

    // ---- workspace bump allocator (~68 MB total) ----
    char* wsbase = (char*)d_ws;
    size_t off = 0;
    auto alloc = [&](size_t bytes) -> void* {
        off = (off + 255) & ~(size_t)255;
        void* p = wsbase + off;
        off += bytes;
        return p;
    };
    // Shared 41 MB region, reused in stream order:
    //   visual keys -> audio keys -> (W_ih || W_hh) -> W_out
    __bf16* R      = (__bf16*)alloc(2ull * VOCABV * HH);          // 41 MB
    __bf16* Xbf    = R;                                           // [B*T, E] keys
    __bf16* Wihbf  = R;                                           // [4H, 1536]
    __bf16* Whhbf  = R + (size_t)4 * HH * (CTXC + INFEAT);        // [4H, H]
    __bf16* Woutbf = R;                                           // [VOCAB, H]

    __bf16* hsbf   = (__bf16*)alloc(2ull * BB * LL * HH);
    __bf16* Wv1bf  = (__bf16*)alloc(2ull * EVv * EVv);
    __bf16* Wa1bf  = (__bf16*)alloc(2ull * EAa * EAa);
    __bf16* Wh1bf  = (__bf16*)alloc(2ull * HH * HH);
    __bf16* Wc1bf  = (__bf16*)alloc(2ull * CTXC * CTXC);
    __bf16* csbf   = (__bf16*)alloc(2ull * BB * 3 * CTXC);
    __bf16* qbf    = (__bf16*)alloc(2ull * BB * CTXC);
    __bf16* Wv0bf  = (__bf16*)alloc(2ull * EVv * CTXC);
    __bf16* Wa0bf  = (__bf16*)alloc(2ull * EAa * CTXC);
    __bf16* Wh0bf  = (__bf16*)alloc(2ull * HH * CTXC);
    __bf16* Wc0bf  = (__bf16*)alloc(2ull * CTXC * HH);
    __bf16* Wacbf  = (__bf16*)alloc(2ull * CTXC * EAa);
    __bf16* Wvcbf  = (__bf16*)alloc(2ull * CTXC * EVv);
    __bf16* Whcbf  = (__bf16*)alloc(2ull * CTXC * HH);
    __bf16* hidbf  = (__bf16*)alloc(2ull * BB * HH);
    __bf16* ctxvbf = (__bf16*)alloc(2ull * BB * EVv);
    __bf16* ctxabf = (__bf16*)alloc(2ull * BB * EAa);
    __bf16* ctxhbf = (__bf16*)alloc(2ull * BB * HH);
    __bf16* xcatbf = (__bf16*)alloc(2ull * BB * (CTXC + INFEAT));
    __bf16* hnbf   = (__bf16*)alloc(2ull * BB * HH);
    float* v1v  = (float*)alloc(4ull * BB * EVv);
    float* v1a  = (float*)alloc(4ull * BB * EAa);
    float* v1h  = (float*)alloc(4ull * BB * HH);
    float* v1f  = (float*)alloc(4ull * BB * CTXC);
    float* scv  = (float*)alloc(4ull * BB * TVv);
    float* sca  = (float*)alloc(4ull * BB * TAa);
    float* sch  = (float*)alloc(4ull * BB * LL);
    float* scf  = (float*)alloc(4ull * BB * 3);
    float* ctxv = (float*)alloc(4ull * BB * EVv);
    float* ctxa = (float*)alloc(4ull * BB * EAa);
    float* ctxh = (float*)alloc(4ull * BB * HH);
    float* cs   = (float*)alloc(4ull * BB * 3 * CTXC);
    float* gates= (float*)alloc(4ull * BB * 4 * HH);

    // ---- output partition ----
    float* out    = (float*)d_out;
    float* logits = out;
    float* h_out  = out + (size_t)BB * VOCABV;
    float* c_out  = h_out + (size_t)BB * HH;
    float* fctx   = c_out + (size_t)BB * HH;

    const float* hidden = hstates + (size_t)(LL - 1) * BB * HH;   // [B,H]

    auto blk = [](long n) -> int {
        long b = (n / 8 + 255) / 256;
        return (int)(b > 4096 ? 4096 : b);
    };
    #define CVT(src, dst, n) cvt_f32_bf16<<<blk(n), 256, 0, stream>>>((src), (dst), (long)(n))

    // ---- persistent bf16 conversions ----
    CVT(Wv1, Wv1bf, (long)EVv * EVv);
    CVT(Wa1, Wa1bf, (long)EAa * EAa);
    CVT(Wh1, Wh1bf, (long)HH * HH);
    CVT(Wc1, Wc1bf, (long)CTXC * CTXC);
    CVT(ctxq, qbf, (long)BB * CTXC);
    CVT(Wv0, Wv0bf, (long)EVv * CTXC);
    CVT(Wa0, Wa0bf, (long)EAa * CTXC);
    CVT(Wh0, Wh0bf, (long)HH * CTXC);
    CVT(Wc0, Wc0bf, (long)CTXC * HH);
    CVT(Wac, Wacbf, (long)CTXC * EAa);
    CVT(Wvc, Wvcbf, (long)CTXC * EVv);
    CVT(Whc, Whcbf, (long)CTXC * HH);
    CVT(hidden, hidbf, (long)BB * HH);
    cvt_hs<<<LL * BB, 128, 0, stream>>>(hstates, hsbf);

    // ---- visual attention ----
    CVT(visual, Xbf, (long)BB * TVv * EVv);
    gemm_bf16_wmma<<<dim3(1, EVv / 128), 256, 0, stream>>>(qbf, CTXC, Wv0bf, CTXC,
        nullptr, nullptr, v1v, EVv, BB, EVv, CTXC, 0);
    attn_scores_wmma<<<(BB * TVv) / 128, 256, 0, stream>>>(Xbf, Wv1bf, v1v, Wv2, scv,
        BB * TVv, TVv, EVv, EVv);
    softmax_ctx<<<BB, 256, 0, stream>>>(scv, visual, (long)TVv * EVv, (long)EVv, ctxv, TVv, EVv);
    CVT(ctxv, ctxvbf, (long)BB * EVv);

    // ---- audio attention (Xbf reused; stream serializes) ----
    CVT(audio, Xbf, (long)BB * TAa * EAa);
    gemm_bf16_wmma<<<dim3(1, EAa / 128), 256, 0, stream>>>(qbf, CTXC, Wa0bf, CTXC,
        nullptr, nullptr, v1a, EAa, BB, EAa, CTXC, 0);
    attn_scores_wmma<<<(BB * TAa) / 128, 256, 0, stream>>>(Xbf, Wa1bf, v1a, Wa2, sca,
        BB * TAa, TAa, EAa, EAa);
    softmax_ctx<<<BB, 256, 0, stream>>>(sca, audio, (long)TAa * EAa, (long)EAa, ctxa, TAa, EAa);
    CVT(ctxa, ctxabf, (long)BB * EAa);

    // ---- hidden-state attention ([L,B,H] via strides) ----
    gemm_bf16_wmma<<<dim3(1, HH / 128), 256, 0, stream>>>(qbf, CTXC, Wh0bf, CTXC,
        nullptr, nullptr, v1h, HH, BB, HH, CTXC, 0);
    attn_scores_wmma<<<(BB * LL) / 128, 256, 0, stream>>>(hsbf, Wh1bf, v1h, Wh2, sch,
        BB * LL, LL, HH, HH);
    softmax_ctx<<<BB, 256, 0, stream>>>(sch, hstates, (long)HH, (long)BB * HH, ctxh, LL, HH);
    CVT(ctxh, ctxhbf, (long)BB * HH);

    // ---- R region now free: stage LSTM weights ----
    CVT(W_ih, Wihbf, (long)4 * HH * (CTXC + INFEAT));
    CVT(W_hh, Whhbf, (long)4 * HH * HH);

    // ---- context stack [B,3,CTX]: audio, visual, hidden projections ----
    gemm_bf16_wmma<<<dim3(1, CTXC / 128), 256, 0, stream>>>(ctxabf, EAa, Wacbf, EAa,
        bac, nullptr, cs + 0 * CTXC, 3 * CTXC, BB, CTXC, EAa, 0);
    gemm_bf16_wmma<<<dim3(1, CTXC / 128), 256, 0, stream>>>(ctxvbf, EVv, Wvcbf, EVv,
        bvc, nullptr, cs + 1 * CTXC, 3 * CTXC, BB, CTXC, EVv, 0);
    gemm_bf16_wmma<<<dim3(1, CTXC / 128), 256, 0, stream>>>(ctxhbf, HH, Whcbf, HH,
        bhc, nullptr, cs + 2 * CTXC, 3 * CTXC, BB, CTXC, HH, 0);
    CVT(cs, csbf, (long)BB * 3 * CTXC);

    // ---- final attention over T=3 context stack ----
    gemm_bf16_wmma<<<dim3(1, CTXC / 128), 256, 0, stream>>>(hidbf, HH, Wc0bf, HH,
        nullptr, nullptr, v1f, CTXC, BB, CTXC, HH, 0);
    attn_scores_wmma<<<(BB * 3 + 127) / 128, 256, 0, stream>>>(csbf, Wc1bf, v1f, Wc2, scf,
        BB * 3, 3, CTXC, CTXC);
    softmax_ctx<<<BB, 256, 0, stream>>>(scf, cs, (long)3 * CTXC, (long)CTXC, fctx, 3, CTXC);

    // ---- x = [final_ctx, input] (bf16); gates = x@W_ih^T + hidden@W_hh^T + biases ----
    cvt2d_f32_bf16<<<BB, 64, 0, stream>>>(fctx, CTXC, xcatbf, CTXC + INFEAT, CTXC);
    cvt2d_f32_bf16<<<BB, 128, 0, stream>>>(input, INFEAT, xcatbf + CTXC, CTXC + INFEAT, INFEAT);
    gemm_bf16_wmma<<<dim3(1, (4 * HH) / 128), 256, 0, stream>>>(xcatbf, CTXC + INFEAT,
        Wihbf, CTXC + INFEAT, b_ih, b_hh, gates, 4 * HH, BB, 4 * HH, CTXC + INFEAT, 0);
    gemm_bf16_wmma<<<dim3(1, (4 * HH) / 128), 256, 0, stream>>>(hidbf, HH,
        Whhbf, HH, nullptr, nullptr, gates, 4 * HH, BB, 4 * HH, HH, 1);

    // ---- LSTM cell (writes h_new/c_new into d_out; bf16 copy for logits) ----
    lstm_cell<<<BB, 256, 0, stream>>>(gates, cell, h_out, c_out, hnbf);

    // ---- logits = h_new @ W_out^T + b_out (R region reused for W_out bf16) ----
    CVT(W_out, Woutbf, (long)VOCABV * HH);
    gemm_bf16_wmma<<<dim3(1, (VOCABV + 127) / 128), 256, 0, stream>>>(hnbf, HH,
        Woutbf, HH, b_out, nullptr, logits, VOCABV, BB, VOCABV, HH, 0);
}
